// Net_44212393345588
// MI455X (gfx1250) — matrix operand order repaired and checked by reading, not compiled, not run on hardware
//
#include <hip/hip_runtime.h>
#include <hip/hip_bf16.h>

typedef _Float16 half_t;
typedef __attribute__((ext_vector_type(16))) _Float16 v16h;
typedef __attribute__((ext_vector_type(8)))  _Float16 v8h;
typedef __attribute__((ext_vector_type(8)))  float    v8f;

#define NN   2048          // nodes
#define HH   64            // hidden
#define BB   16            // batch
#define TT   12            // encoder steps
#define FUT  12            // decoder steps
#define FEAT 66            // IN + H
#define CPAD 1088          // 16*66=1056 cols padded to multiple of 64
#define KW   224           // K = 3*66 = 198 padded to multiple of 32
#define KSTEP 64           // k-panel width staged per TDM transfer

#ifndef USE_TDM
#if __has_builtin(__builtin_amdgcn_tensor_load_to_lds)
#define USE_TDM 1
#else
#define USE_TDM 0
#endif
#endif

#if USE_TDM
typedef __attribute__((ext_vector_type(4))) unsigned int v4u;
typedef __attribute__((ext_vector_type(4))) int          v4i;
typedef __attribute__((ext_vector_type(8))) int          v8i;

// TDM: DMA a 2-D tile (rows x tileW f16, row stride strideElems) from global
// into contiguous LDS at lds_addr. Descriptor per ISA 8.3/8.4:
//   group0: count=1 | lds_addr | global_addr[56:0] | type=2
//   group1: data_size=1(2B), tensor_dim0=stride, tensor_dim1=rows,
//           tile_dim0=tileW, tile_dim1=rows, tensor_dim0_stride=stride
__device__ __forceinline__ void tdm_load_2d(const void* gsrc, unsigned lds_addr,
                                            unsigned rows, unsigned strideElems,
                                            unsigned tileW) {
    unsigned long long ga = (unsigned long long)(uintptr_t)gsrc;
    v4u g0;
    g0[0] = 1u;                                              // count=1 (valid)
    g0[1] = lds_addr;                                        // LDS byte address
    g0[2] = (unsigned)ga;                                    // global addr lo
    g0[3] = ((unsigned)(ga >> 32) & 0x01FFFFFFu) | (2u << 30); // addr[56:32]|type=2
    v8i g1;
    g1[0] = (int)(1u << 16);                                 // data_size = 2 bytes
    g1[1] = (int)((strideElems & 0xFFFFu) << 16);            // tensor_dim0 lo16
    g1[2] = (int)(((strideElems >> 16) & 0xFFFFu) | ((rows & 0xFFFFu) << 16));
    g1[3] = (int)(((rows >> 16) & 0xFFFFu) | (tileW << 16)); // tile_dim0
    g1[4] = (int)rows;                                       // tile_dim1 = rows
    g1[5] = (int)strideElems;                                // dim0 stride lo32
    g1[6] = 0;
    g1[7] = 0;
    v4i z4 = {0, 0, 0, 0};
#if defined(__clang_major__) && __clang_major__ >= 23
    v8i z8 = {0, 0, 0, 0, 0, 0, 0, 0};
    __builtin_amdgcn_tensor_load_to_lds(g0, g1, z4, z4, z8, 0);
#else
    __builtin_amdgcn_tensor_load_to_lds(g0, g1, z4, z4, 0);
#endif
}
#endif // USE_TDM

// ---------------------------------------------------------------------------
// Heavy GEMM: Out[c][m] (transposed store) = sum_k L16[m][k] * X[c][k]
//   mode==0 : Out = L*X                  (T1 = L @ comb)
//   mode==1 : Out = 2*(L*X) - T0        (T2 = 2 L T1 - T0)
// Workgroup (8 waves) owns a 128m x 64c macro tile. A (128x64) and B (64x64)
// k-panels are double-buffered in LDS; the Tensor Data Mover prefetches both
// panels one k-step ahead (TENSORcnt). One barrier per 8 WMMAs.
// ---------------------------------------------------------------------------
__global__ void gemmL_kernel(const half_t* __restrict__ L16,
                             const half_t* __restrict__ X,
                             const half_t* __restrict__ T0,
                             half_t* __restrict__ Out,
                             int mode) {
#if USE_TDM
    __shared__ half_t As[2][128 * KSTEP];  // 32 KB double-buffered A panel
#endif
    __shared__ half_t Bs[2][64 * KSTEP];   // 16 KB double-buffered B panel

    int mBlk = blockIdx.x / (CPAD / 64);   // 0..15  (128 rows each)
    int cBlk = blockIdx.x % (CPAD / 64);   // 0..16  (64 cols each)
    int c0 = cBlk * 64;
    int tid  = threadIdx.x;
    int w    = tid >> 5;
    int lane = tid & 31;
    int lh = lane & 15;
    int hi = (lane >> 4) & 1;
    int m0 = mBlk * 128 + w * 16;

    v8f acc[4];
#pragma unroll
    for (int j = 0; j < 4; ++j)
#pragma unroll
        for (int r = 0; r < 8; ++r) acc[j][r] = 0.0f;

#if USE_TDM
    // LDS byte offsets: flat LDS addresses truncate to addr[31:0] (ISA 10.2)
    unsigned ldsA[2] = { (unsigned)(unsigned long long)(void*)&As[0][0],
                         (unsigned)(unsigned long long)(void*)&As[1][0] };
    unsigned ldsB[2] = { (unsigned)(unsigned long long)(void*)&Bs[0][0],
                         (unsigned)(unsigned long long)(void*)&Bs[1][0] };
    const half_t* gA = L16 + (size_t)(mBlk * 128) * NN;   // + k
    const half_t* gB = X   + (size_t)c0 * NN;             // + k

    if (w == 0) {
        tdm_load_2d(gA, ldsA[0], 128, NN, KSTEP);
        tdm_load_2d(gB, ldsB[0], 64, NN, KSTEP);
        __builtin_amdgcn_s_wait_tensorcnt(0);
    }
    __syncthreads();

    for (int k = 0; k < NN; k += KSTEP) {
        int cur = (k >> 6) & 1;
        if (w == 0 && k + KSTEP < NN) {
            tdm_load_2d(gA + k + KSTEP, ldsA[cur ^ 1], 128, NN, KSTEP);
            tdm_load_2d(gB + k + KSTEP, ldsB[cur ^ 1], 64, NN, KSTEP);
        }
#pragma unroll
        for (int s = 0; s < KSTEP; s += 32) {
            // A fragment: lanes 0-15 K 0-7 & 16-23, lanes 16-31 K 8-15 & 24-31
            union { v16h v; v8h h[2]; } ua;
            const half_t* ap = &As[cur][(w * 16 + lh) * KSTEP + s + (hi ? 8 : 0)];
            ua.h[0] = *(const v8h*)(ap);
            ua.h[1] = *(const v8h*)(ap + 16);
            v16h a = ua.v;
            v16h bf[4];
#pragma unroll
            for (int j = 0; j < 4; ++j)
            {
                union { v16h v; v8h h[2]; } ub;
                const half_t* bp = &Bs[cur][(j * 16 + lh) * KSTEP + s + (hi ? 8 : 0)];
                ub.h[0] = *(const v8h*)(bp);
                ub.h[1] = *(const v8h*)(bp + 16);
                bf[j] = ub.v;
            }
#pragma unroll
            for (int j = 0; j < 4; ++j)
                acc[j] = __builtin_amdgcn_wmma_f32_16x16x32_f16(
                    false, a, false, bf[j], (short)0, acc[j], false, false);
        }
        if (w == 0) __builtin_amdgcn_s_wait_tensorcnt(0);
        __syncthreads();
    }
#else
    // Fallback: cooperative B staging through VGPRs, A streamed from global.
    int sCol = tid >> 2;
    int sSeg = (tid & 3) * 8;
    const half_t* sSrc = X + (size_t)(c0 + sCol) * NN + sSeg;
    const half_t* arow = L16 + (size_t)(m0 + lh) * NN;

    *(v8h*)&Bs[0][sCol * KSTEP + sSeg]      = *(const v8h*)(sSrc);
    *(v8h*)&Bs[0][sCol * KSTEP + sSeg + 32] = *(const v8h*)(sSrc + 32);
    __syncthreads();

    for (int k = 0; k < NN; k += KSTEP) {
        int cur = (k >> 6) & 1;
        if (k + KSTEP < NN) {
            *(v8h*)&Bs[cur ^ 1][sCol * KSTEP + sSeg]      = *(const v8h*)(sSrc + k + KSTEP);
            *(v8h*)&Bs[cur ^ 1][sCol * KSTEP + sSeg + 32] = *(const v8h*)(sSrc + k + KSTEP + 32);
        }
#pragma unroll
        for (int s = 0; s < KSTEP; s += 32) {
            union { v16h v; v8h h[2]; } ua;
            const half_t* ap = arow + k + s + (hi ? 8 : 0);
            ua.h[0] = *(const v8h*)(ap);
            ua.h[1] = *(const v8h*)(ap + 16);
            v16h a = ua.v;
#pragma unroll
            for (int j = 0; j < 4; ++j) {
                union { v16h v; v8h h[2]; } ub;
                const half_t* bp = &Bs[cur][(j * 16 + lh) * KSTEP + s + (hi ? 8 : 0)];
                ub.h[0] = *(const v8h*)(bp);
                ub.h[1] = *(const v8h*)(bp + 16);
                v16h bfj = ub.v;
                acc[j] = __builtin_amdgcn_wmma_f32_16x16x32_f16(
                    false, a, false, bfj, (short)0, acc[j], false, false);
            }
        }
        __syncthreads();
    }
#endif

    // D tile: VGPR r, lane l<16 -> (M=r, N=l); l>=16 -> (M=r+8, N=l-16).
    int mrow = m0 + (hi ? 8 : 0);
#pragma unroll
    for (int j = 0; j < 4; ++j) {
        size_t idx = (size_t)(c0 + j * 16 + lh) * NN + mrow;
        v8h o;
        if (mode) {
            v8h t0 = *(const v8h*)(T0 + idx);
#pragma unroll
            for (int r = 0; r < 8; ++r)
                o[r] = (half_t)(2.0f * acc[j][r] - (float)t0[r]);
        } else {
#pragma unroll
            for (int r = 0; r < 8; ++r)
                o[r] = (half_t)acc[j][r];
        }
        *(v8h*)(Out + idx) = o;
    }
}

// ---------------------------------------------------------------------------
// Fused gate GEMM + LSTM pointwise + output projection.
// One workgroup owns (batch b, 64-node block): it computes all 256 gate rows
//   G[g][n] = sum_k WT[g][k] * cheb[k][(b,n)]      (WMMA, cheb panel in LDS)
// then transposes the accumulators through LDS and applies
//   c2 = sig(f)*c + sig(i)*tanh(g); h2 = sig(o)*tanh(c2)
// writing c2 (f32) and h2 (f16, straight into the T0 comb rows). For
// step >= 0 it also computes x = tanh(h2 @ W0 + b0), writes the x comb rows
// and output slot `step`. Blocks touch disjoint 64-col slices -> race-free.
// LDS phases alias: [Cs 28KB] then [Gs 32KB | Hs 8KB].
// ---------------------------------------------------------------------------
__global__ void gates_lstm_kernel(const half_t* __restrict__ WT,
                                  const half_t* __restrict__ T0,
                                  const half_t* __restrict__ T1,
                                  const half_t* __restrict__ T2,
                                  const float* __restrict__ bc,
                                  float* __restrict__ Ct,
                                  half_t* __restrict__ T0w,
                                  const float* __restrict__ W0,
                                  const float* __restrict__ b0,
                                  float* __restrict__ out, int step) {
    __shared__ __align__(32) char smem[40 * 1024];
    half_t* Cs = (half_t*)smem;                   // phase 1: 64*KW  = 28 KB
    half_t* Gs = (half_t*)smem;                   // phase 2: 256*64 = 32 KB
    half_t* Hs = (half_t*)(smem + 32 * 1024);     // phase 2: 64*64  =  8 KB

    int b    = blockIdx.x >> 5;           // 16 batches
    int nBlk = blockIdx.x & 31;           // 32 blocks of 64 nodes
    int n0 = nBlk * 64;
    int tid = threadIdx.x;

    // ---- phase 1: stage cheb panel (column-major) and run the gate GEMM ----
    if (tid < KW) {
        int k = tid;
        if (k < 198) {
            int t = (k < 66) ? 0 : ((k < 132) ? 1 : 2);
            int f = k - t * 66;
            const half_t* src = ((t == 0) ? T0 : (t == 1) ? T1 : T2)
                                + (size_t)(b * FEAT + f) * NN + n0;
#pragma unroll 8
            for (int col = 0; col < 64; ++col) Cs[col * KW + k] = src[col];
        } else {
#pragma unroll 8
            for (int col = 0; col < 64; ++col) Cs[col * KW + k] = (half_t)0.0f;
        }
    }
    __syncthreads();

    int w    = tid >> 5;
    int lane = tid & 31;
    int lh = lane & 15;
    int hi = (lane >> 4) & 1;
    int gt0 = w * 2;                      // wave's two gate tiles (rows w*32..w*32+31)

    v8f acc[2][4];
#pragma unroll
    for (int gi = 0; gi < 2; ++gi)
#pragma unroll
        for (int j = 0; j < 4; ++j)
#pragma unroll
            for (int r = 0; r < 8; ++r) acc[gi][j][r] = 0.0f;

#pragma unroll
    for (int kt = 0; kt < 7; ++kt) {
        int k0 = kt * 32;
        v16h a[2];
#pragma unroll
        for (int gi = 0; gi < 2; ++gi) {
            union { v16h v; v8h h[2]; } ua;
            const half_t* ap = WT + (size_t)((gt0 + gi) * 16 + lh) * KW
                               + k0 + (hi ? 8 : 0);
            ua.h[0] = *(const v8h*)(ap);
            ua.h[1] = *(const v8h*)(ap + 16);
            a[gi] = ua.v;
        }
#pragma unroll
        for (int j = 0; j < 4; ++j) {
            union { v16h v; v8h h[2]; } ub;
            const half_t* bp = &Cs[(j * 16 + lh) * KW + k0 + (hi ? 8 : 0)];
            ub.h[0] = *(const v8h*)(bp);
            ub.h[1] = *(const v8h*)(bp + 16);
            v16h bf = ub.v;
#pragma unroll
            for (int gi = 0; gi < 2; ++gi)
                acc[gi][j] = __builtin_amdgcn_wmma_f32_16x16x32_f16(
                    false, a[gi], false, bf, (short)0, acc[gi][j], false, false);
        }
    }
    __syncthreads();   // Cs dead; safe to overwrite with Gs

    // ---- phase 2: transpose accumulators into Gs[g][nLoc] ----
#pragma unroll
    for (int gi = 0; gi < 2; ++gi)
#pragma unroll
        for (int j = 0; j < 4; ++j) {
            int nLoc = j * 16 + lh;
#pragma unroll
            for (int r = 0; r < 8; ++r) {
                int g = (gt0 + gi) * 16 + r + (hi ? 8 : 0);
                Gs[g * 64 + nLoc] = (half_t)acc[gi][j][r];
            }
        }
    __syncthreads();

    // ---- LSTM gating: 4096 (j, nLoc) pairs, 16 per thread ----
#pragma unroll
    for (int i = 0; i < 16; ++i) {
        int e = tid + 256 * i;
        int j = e >> 6;
        int nLoc = e & 63;
        float vi = (float)Gs[(0   + j) * 64 + nLoc] + bc[j];
        float vf = (float)Gs[(64  + j) * 64 + nLoc] + bc[64 + j];
        float vo = (float)Gs[(128 + j) * 64 + nLoc] + bc[128 + j];
        float vg = (float)Gs[(192 + j) * 64 + nLoc] + bc[192 + j];
        size_t cidx = (size_t)(b * HH + j) * NN + n0 + nLoc;
        float c_old = Ct[cidx];
        float si = 1.0f / (1.0f + __expf(-vi));
        float sf = 1.0f / (1.0f + __expf(-vf));
        float so = 1.0f / (1.0f + __expf(-vo));
        float c2 = sf * c_old + si * tanhf(vg);
        float h2 = so * tanhf(c2);
        Ct[cidx] = c2;
        T0w[(size_t)(b * FEAT + 2 + j) * NN + n0 + nLoc] = (half_t)h2;
        Hs[j * 64 + nLoc] = (half_t)h2;
    }

    // ---- output projection x = tanh(h @ W0 + b0) (decoder / final encoder) ----
    if (step >= 0) {
        __syncthreads();
        if (tid < 64) {
            int n = n0 + tid;
            float a0 = b0[0], a1 = b0[1];
#pragma unroll 8
            for (int j = 0; j < HH; ++j) {
                float h = (float)Hs[j * 64 + tid];
                a0 += h * W0[j * 2 + 0];
                a1 += h * W0[j * 2 + 1];
            }
            float x0 = tanhf(a0), x1 = tanhf(a1);
            T0w[(size_t)(b * FEAT + 0) * NN + n] = (half_t)x0;
            T0w[(size_t)(b * FEAT + 1) * NN + n] = (half_t)x1;
            size_t o = (((size_t)b * FUT + step) * NN + n) * 2;
            out[o]     = x0;
            out[o + 1] = x1;
        }
    }
}

// Encoder: scatter input step t into the x feature rows of T0.
__global__ void fillx_kernel(const float* __restrict__ in,
                             half_t* __restrict__ T0buf, int t) {
    int tid = blockIdx.x * blockDim.x + threadIdx.x;
    if (tid >= BB * NN) return;
    int n = tid & (NN - 1);
    int b = tid >> 11;
    size_t ibase = (((size_t)b * TT + t) * NN + n) * 2;
    T0buf[(size_t)(b * FEAT + 0) * NN + n] = (half_t)in[ibase];
    T0buf[(size_t)(b * FEAT + 1) * NN + n] = (half_t)in[ibase + 1];
}

__global__ void cvt_half_kernel(const float* __restrict__ src,
                                half_t* __restrict__ dst, int nElem) {
    int tid = blockIdx.x * blockDim.x + threadIdx.x;
    if (tid < nElem) dst[tid] = (half_t)src[tid];
}

// WT[g][k] = W_cell[k][g] for k<198, else 0 (K padded to 224).
__global__ void buildWT_kernel(const float* __restrict__ Wc,
                               half_t* __restrict__ WT) {
    int tid = blockIdx.x * blockDim.x + threadIdx.x;
    if (tid >= 256 * KW) return;
    int g = tid / KW;
    int k = tid % KW;
    WT[tid] = (k < 198) ? (half_t)Wc[k * 256 + g] : (half_t)0.0f;
}

__global__ void zero_half_kernel(half_t* p, int n) {
    int tid = blockIdx.x * blockDim.x + threadIdx.x;
    if (tid < n) p[tid] = (half_t)0.0f;
}
__global__ void zero_float_kernel(float* p, int n) {
    int tid = blockIdx.x * blockDim.x + threadIdx.x;
    if (tid < n) p[tid] = 0.0f;
}

extern "C" void kernel_launch(void* const* d_in, const int* in_sizes, int n_in,
                              void* d_out, int out_size, void* d_ws, size_t ws_size,
                              hipStream_t stream) {
    const float* in_x = (const float*)d_in[0];   // [B,T,N,2]
    const float* Lf   = (const float*)d_in[1];   // [N,N]
    const float* Wc   = (const float*)d_in[2];   // [198,256]
    const float* bc   = (const float*)d_in[3];   // [256]
    const float* W0   = (const float*)d_in[4];   // [64,2]
    const float* b0   = (const float*)d_in[5];   // [2]
    float* out = (float*)d_out;                  // [B,12,N,2]

    // Workspace carve-out (all slices multiple-of-4KB aligned).
    char* w = (char*)d_ws;
    half_t* L16 = (half_t*)w; w += (size_t)NN * NN * sizeof(half_t);
    half_t* T0  = (half_t*)w; w += (size_t)CPAD * NN * sizeof(half_t);
    half_t* T1  = (half_t*)w; w += (size_t)CPAD * NN * sizeof(half_t);
    half_t* T2  = (half_t*)w; w += (size_t)CPAD * NN * sizeof(half_t);
    half_t* WT  = (half_t*)w; w += (size_t)256 * KW * sizeof(half_t);
    float*  Ct  = (float*)w;  w += (size_t)BB * HH * NN * sizeof(float);

    // ---- init (every call: deterministic; resets h, c, pad rows) ----
    {
        int nL = NN * NN;
        cvt_half_kernel<<<(nL + 255) / 256, 256, 0, stream>>>(Lf, L16, nL);
        buildWT_kernel<<<(256 * KW + 255) / 256, 256, 0, stream>>>(Wc, WT);
        int nT = CPAD * NN;
        zero_half_kernel<<<(nT + 255) / 256, 256, 0, stream>>>(T0, nT);
        int nC = BB * HH * NN;
        zero_float_kernel<<<(nC + 255) / 256, 256, 0, stream>>>(Ct, nC);
    }

    const int GEMM_BLOCKS = 16 * (CPAD / 64);   // 272 workgroups (128m x 64c tiles)
    const int GATE_BLOCKS = BB * 32;            // 512 workgroups (256g x 64n tiles)
    const int XN_BLOCKS   = (BB * NN) / 256;

    auto cell = [&](int step) {
        gemmL_kernel<<<GEMM_BLOCKS, 256, 0, stream>>>(L16, T0, (const half_t*)nullptr, T1, 0);
        gemmL_kernel<<<GEMM_BLOCKS, 256, 0, stream>>>(L16, T1, T0, T2, 1);
        gates_lstm_kernel<<<GATE_BLOCKS, 256, 0, stream>>>(WT, T0, T1, T2, bc, Ct,
                                                           T0, W0, b0, out, step);
    };

    // Encoder: 12 steps (last one also emits output slot 0)
    for (int t = 0; t < TT; ++t) {
        fillx_kernel<<<XN_BLOCKS, 256, 0, stream>>>(in_x, T0, t);
        cell(t == TT - 1 ? 0 : -1);
    }
    // Decoder: 11 more steps emit slots 1..11
    for (int s = 1; s < FUT; ++s) {
        cell(s);
    }
}
